// HardTripletLoss_15977278341300
// MI455X (gfx1250) — compile-verified, matrix operand order checked by reading, not run on hardware
//
#include <hip/hip_runtime.h>

#define TRIPLET_MARGIN 0.2f

typedef __bf16 bf16_t;
typedef __attribute__((ext_vector_type(8)))  bf16_t v8bf;
typedef __attribute__((ext_vector_type(16))) bf16_t v16bf;
typedef __attribute__((ext_vector_type(8)))  float  v8f;

// -----------------------------------------------------------------------------
// Kernel 1: per-row prep. One block (128 threads) per row i.
//   - converts x,y rows to bf16 (workspace)
//   - computes yy[i] = sum(y_i^2), gii = x_i . y_i  (full fp32 precision)
//   - writes a[i] = yy[i] - 2*gii + margin, b[i] = yy[i]
//   - row 0 / thread 0 zeroes the global accumulator
// -----------------------------------------------------------------------------
__global__ __launch_bounds__(128) void triplet_prep_kernel(
    const float* __restrict__ x, const float* __restrict__ y,
    bf16_t* __restrict__ xb, bf16_t* __restrict__ yb,
    float* __restrict__ arow, float* __restrict__ bcol,
    float* __restrict__ accum, int D)
{
    const int i = blockIdx.x;
    const int d = threadIdx.x;
    const size_t off = (size_t)i * D + d;

    const float xf = x[off];
    const float yf = y[off];
    xb[off] = (bf16_t)xf;
    yb[off] = (bf16_t)yf;

    __shared__ float s1[128];
    __shared__ float s2[128];
    s1[d] = yf * yf;
    s2[d] = xf * yf;
    __syncthreads();
    #pragma unroll
    for (int o = 64; o > 0; o >>= 1) {
        if (d < o) { s1[d] += s1[d + o]; s2[d] += s2[d + o]; }
        __syncthreads();
    }
    if (d == 0) {
        const float yy  = s1[0];
        const float gii = s2[0];
        bcol[i] = yy;
        arow[i] = yy - 2.0f * gii + TRIPLET_MARGIN;
        if (i == 0) *accum = 0.0f;
    }
}

// -----------------------------------------------------------------------------
// Kernel 2: fused bf16 WMMA GEMM (G = x . y^T) + hinge-loss reduction.
// Block = 256 threads = 8 wave32 waves. Block tile: 256(M) x 128(N), K = 128.
// Wave w owns rows [blockIdx.y*256 + w*32, +32) x all 128 cols:
//   2 M-subtiles x 8 N-subtiles = 16 C tiles (v8f each), 64 WMMAs per wave.
// Per k-step: 4 A-frag loads + 16 B-frag loads feed 16 WMMAs; all loads are
// hoisted into distinct register buffers so the scheduler can overlap them
// with the WMMA chain instead of s_wait_loadcnt 0 before every WMMA.
// -----------------------------------------------------------------------------
__global__ __launch_bounds__(256) void triplet_gemm_loss_kernel(
    const bf16_t* __restrict__ xb, const bf16_t* __restrict__ yb,
    const float* __restrict__ arow, const float* __restrict__ bcol,
    float* __restrict__ accum)
{
    const int D    = 128;
    const int wave = threadIdx.x >> 5;
    const int lane = threadIdx.x & 31;
    const int lm   = lane & 15;   // M-row (A) / N-col (B) within a 16-tile
    const int hs   = lane >> 4;   // half-select for the two 16B K-runs

    const int row_base = blockIdx.y * 256 + wave * 32;
    const int col_base = blockIdx.x * 128;

    const bf16_t* aptr0 = xb + (size_t)(row_base + lm) * D;
    const bf16_t* aptr1 = xb + (size_t)(row_base + 16 + lm) * D;

    v8f acc[2][8] = {};

    union Frag { v16bf v; v8bf h[2]; };

    #pragma unroll
    for (int ks = 0; ks < 4; ++ks) {
        const int k0 = ks * 32;

        // A fragments: 16x32 bf16 each; lane (l&15)=row, two 16B K-runs.
        Frag A0, A1;
        A0.h[0] = *(const v8bf*)(aptr0 + k0 + hs * 8);
        A0.h[1] = *(const v8bf*)(aptr0 + k0 + 16 + hs * 8);
        A1.h[0] = *(const v8bf*)(aptr1 + k0 + hs * 8);
        A1.h[1] = *(const v8bf*)(aptr1 + k0 + 16 + hs * 8);

        // All 8 B fragments up-front (C = A.B^T => same fetch pattern on y rows)
        Frag B[8];
        #pragma unroll
        for (int t = 0; t < 8; ++t) {
            const bf16_t* bptr = yb + (size_t)(col_base + t * 16 + lm) * D;
            B[t].h[0] = *(const v8bf*)(bptr + k0 + hs * 8);
            B[t].h[1] = *(const v8bf*)(bptr + k0 + 16 + hs * 8);
        }

        // 16 WMMAs; each B fragment feeds two (A0, A1).
        #pragma unroll
        for (int t = 0; t < 8; ++t) {
            acc[0][t] = __builtin_amdgcn_wmma_f32_16x16x32_bf16(
                false, A0.v, false, B[t].v, (short)0, acc[0][t], false, false);
            acc[1][t] = __builtin_amdgcn_wmma_f32_16x16x32_bf16(
                false, A1.v, false, B[t].v, (short)0, acc[1][t], false, false);
        }
    }

    // Epilogue: C layout (16x16 f32): VGPR q, lane l -> M = q + 8*(l>=16), N = l&15
    float part = 0.0f;
    #pragma unroll
    for (int s = 0; s < 2; ++s) {
        #pragma unroll
        for (int t = 0; t < 8; ++t) {
            const int j    = col_base + t * 16 + lm;
            const float bj = bcol[j];
            #pragma unroll
            for (int q = 0; q < 8; ++q) {
                const int i = row_base + s * 16 + hs * 8 + q;
                float v = fmaxf(arow[i] - bj + 2.0f * acc[s][t][q], 0.0f);
                if (i == j) v = 0.0f;   // zero the diagonal
                part += v;
            }
        }
    }

    // Block reduction + single atomic per block
    __shared__ float red[256];
    red[threadIdx.x] = part;
    __syncthreads();
    #pragma unroll
    for (int o = 128; o > 0; o >>= 1) {
        if ((int)threadIdx.x < o) red[threadIdx.x] += red[threadIdx.x + o];
        __syncthreads();
    }
    if (threadIdx.x == 0) atomicAdd(accum, red[0]);
}

// -----------------------------------------------------------------------------
// Kernel 3: finalize -> mean over N*N
// -----------------------------------------------------------------------------
__global__ void triplet_finalize_kernel(const float* __restrict__ accum,
                                        float* __restrict__ out, float invNN)
{
    out[0] = accum[0] * invNN;
}

extern "C" void kernel_launch(void* const* d_in, const int* in_sizes, int n_in,
                              void* d_out, int out_size, void* d_ws, size_t ws_size,
                              hipStream_t stream)
{
    (void)n_in; (void)out_size; (void)ws_size;
    const float* x = (const float*)d_in[0];
    const float* y = (const float*)d_in[1];

    const int D = 128;
    const int N = in_sizes[0] / D;   // 8192 for the reference shapes

    // Workspace layout: xb[N*D] bf16 | yb[N*D] bf16 | a[N] f32 | b[N] f32 | accum
    char* ws = (char*)d_ws;
    const size_t bf_bytes = (size_t)N * D * sizeof(unsigned short);
    bf16_t* xb    = (bf16_t*)ws;
    bf16_t* yb    = (bf16_t*)(ws + bf_bytes);
    float*  arow  = (float*)(ws + 2 * bf_bytes);
    float*  bcol  = arow + N;
    float*  accum = bcol + N;

    triplet_prep_kernel<<<N, 128, 0, stream>>>(x, y, xb, yb, arow, bcol, accum, D);

    dim3 grid(N / 128, N / 256);
    triplet_gemm_loss_kernel<<<grid, 256, 0, stream>>>(xb, yb, arow, bcol, accum);

    const float invNN = 1.0f / ((float)N * (float)N);
    triplet_finalize_kernel<<<1, 1, 0, stream>>>(accum, (float*)d_out, invNN);
}
